// SpikeVSAOps_23270132810162
// MI455X (gfx1250) — compile-verified, matrix operand order checked by reading, not run on hardware
//
#include <hip/hip_runtime.h>
#include <hip/hip_bf16.h>
#include <math.h>

// ---------------------------------------------------------------------------
// Resonator decode on MI455X (gfx1250, wave32, WMMA f32_16x16x32_f16)
//   B=8 (padded to M=16), K=1024, DIM=4096, 50 iterations
//   iteration = sims-GEMM (split-K x4, LDS reduce)
//             -> phasor-GEMM with fused in-LDS softmax and rsq-normalized
//                f16 feature output (cos(atan2(im,re)) = re/|z|, sin = im/|z|)
// ---------------------------------------------------------------------------

typedef _Float16 v8h  __attribute__((ext_vector_type(8)));
typedef _Float16 v16h __attribute__((ext_vector_type(16)));
typedef float    v8f  __attribute__((ext_vector_type(8)));

#define DIM_     4096
#define KCB_     1024
#define BROWS_   8
#define FEAT_    8192   // 2*DIM (cos | sin)

// Per-lane A/B fragment load for V_WMMA_F32_16X16X32_F16.
// ISA layout (CDNA5 ISA §7.12.2): lane L holds row/col = L&15, with
// k-halves {khb..khb+7} in VGPR0..3 and {khb+16..khb+23} in VGPR4..7,
// khb = (L>=16) ? 8 : 0.  Two 16-byte contiguous loads per lane.
__device__ __forceinline__ v16h load_frag16(const _Float16* __restrict__ p) {
    const v8h lo = *reinterpret_cast<const v8h*>(p);
    const v8h hi = *reinterpret_cast<const v8h*>(p + 16);
    return __builtin_shufflevector(lo, hi,
        0, 1, 2, 3, 4, 5, 6, 7, 8, 9, 10, 11, 12, 13, 14, 15);
}

// --- prep: f32 codebook angles -> f16 cos/sin in the two layouts we need ---
__global__ __launch_bounds__(256) void k_prep_cb(const float* __restrict__ cb,
                                                 _Float16* __restrict__ cbF,     // [K][8192]  (sims B operand)
                                                 _Float16* __restrict__ cbCosT,  // [DIM][K]   (phasor B operand, re)
                                                 _Float16* __restrict__ cbSinT)  // [DIM][K]   (phasor B operand, im)
{
    const int i = blockIdx.x * 256 + threadIdx.x;      // < K*DIM
    const int k = i >> 12;
    const int d = i & (DIM_ - 1);
    float s, c;
    __sincosf(cb[i], &s, &c);                          // args in [0, 2pi); f16 dest
    cbF[k * FEAT_ + d]        = (_Float16)c;
    cbF[k * FEAT_ + DIM_ + d] = (_Float16)s;
    cbCosT[d * KCB_ + k] = (_Float16)c;
    cbSinT[d * KCB_ + k] = (_Float16)s;
}

// --- init: estF rows 0..7 from superposed angles, rows 8..15 = angle 0 -----
__global__ __launch_bounds__(256) void k_feat_init(const float* __restrict__ sup,
                                                   _Float16* __restrict__ estF)
{
    const int i = blockIdx.x * 256 + threadIdx.x;      // < 16*DIM
    const int m = i >> 12;
    const int d = i & (DIM_ - 1);
    const float ang = (m < BROWS_) ? sup[i] : 0.0f;    // rows >=8 deterministic
    float s, c;
    __sincosf(ang, &s, &c);
    estF[m * FEAT_ + d]        = (_Float16)c;
    estF[m * FEAT_ + DIM_ + d] = (_Float16)s;
}

// --- sims GEMM: (16 x 8192) x (8192 x 1024), split-K x4 + LDS reduce -------
// Block = 8 waves = 2 N-tiles x 4 K-splits. Each wave runs two independent
// WMMA accumulator chains (breaks the RAW chain on the accumulator).
template <int MAXROW>
__global__ __launch_bounds__(256) void k_sims(const _Float16* __restrict__ estF,
                                              const _Float16* __restrict__ cbF,
                                              float* __restrict__ out,
                                              int ldo, float scale)
{
    const int lane  = threadIdx.x & 31;
    const int wave  = threadIdx.x >> 5;
    const int tile  = blockIdx.x * 2 + (wave >> 2);    // 0..63 (N tiles of 16)
    const int split = wave & 3;                        // K split 0..3
    const int rc    = lane & 15;
    const int khb   = (lane >> 4) * 8;

    const _Float16* aBase = estF + rc * FEAT_ + khb;
    const _Float16* bBase = cbF + (size_t)(tile * 16 + rc) * FEAT_ + khb;

    const int kBeg = split * (FEAT_ / 4);
    const int kEnd = kBeg + (FEAT_ / 4);               // 2048 halves per split

    v8f acc0 = {}, acc1 = {};
#pragma unroll 2
    for (int k0 = kBeg; k0 < kEnd; k0 += 64) {
        __builtin_prefetch(bBase + k0 + 512, 0, 3);    // WGP-scope: codebook is cache-resident
        const v16h a0 = load_frag16(aBase + k0);
        const v16h b0 = load_frag16(bBase + k0);
        const v16h a1 = load_frag16(aBase + k0 + 32);
        const v16h b1 = load_frag16(bBase + k0 + 32);
        acc0 = __builtin_amdgcn_wmma_f32_16x16x32_f16(
            false, a0, false, b0, (short)0, acc0, false, false);
        acc1 = __builtin_amdgcn_wmma_f32_16x16x32_f16(
            false, a1, false, b1, (short)0, acc1, false, false);
    }
    const v8f acc = acc0 + acc1;

    // Cross-wave split-K reduction through LDS.
    __shared__ float red[8][256];
#pragma unroll
    for (int r = 0; r < 8; ++r) red[wave][r * 32 + lane] = acc[r];
    __syncthreads();

    if (split == 0) {
        const int wbase = wave;                        // wave is 0 or 4 here
        const int n = tile * 16 + rc;
#pragma unroll
        for (int r = 0; r < 8; ++r) {
            const int m = khb + r;
            if (MAXROW == 16 || m < MAXROW) {
                float v = red[wbase][r * 32 + lane] + red[wbase + 1][r * 32 + lane]
                        + red[wbase + 2][r * 32 + lane] + red[wbase + 3][r * 32 + lane];
                out[m * ldo + n] = v * scale;
            }
        }
    }
}

// --- phasor GEMM with fused softmax -----------------------------------------
// Every block recomputes softmax(logits) (16 x 1024, trivial) into LDS, then
// runs re/im = w @ cb_{cos,sin} with the A operand served from LDS, and
// writes the next f16 feature matrix directly (rsq-normalized phasor).
// One wave per 16-wide D tile (256 tiles), 4 independent WMMA chains.
__global__ __launch_bounds__(256) void k_phasor(const float* __restrict__ logits,
                                                const _Float16* __restrict__ cbCosT,
                                                const _Float16* __restrict__ cbSinT,
                                                _Float16* __restrict__ estF)
{
    __shared__ _Float16 wLDS[16 * KCB_];               // 32 KB softmax weights
    const int lane = threadIdx.x & 31;
    const int wave = threadIdx.x >> 5;

    // ---- fused softmax: wave handles rows 2*wave and 2*wave+1 ----
#pragma unroll
    for (int rr = 0; rr < 2; ++rr) {
        const int row = wave * 2 + rr;
        const float* lr = logits + row * KCB_;
        float vals[32];
        float mx = -INFINITY;
#pragma unroll
        for (int j = 0; j < 32; ++j) {
            vals[j] = lr[lane + 32 * j];
            mx = fmaxf(mx, vals[j]);
        }
#pragma unroll
        for (int s = 16; s > 0; s >>= 1) mx = fmaxf(mx, __shfl_xor(mx, s, 32));
        float sum = 0.0f;
#pragma unroll
        for (int j = 0; j < 32; ++j) {
            vals[j] = __expf(vals[j] - mx);            // v_exp_f32
            sum += vals[j];
        }
#pragma unroll
        for (int s = 16; s > 0; s >>= 1) sum += __shfl_xor(sum, s, 32);
        const float inv = __builtin_amdgcn_rcpf(sum);  // v_rcp_f32 (f16 dest)
#pragma unroll
        for (int j = 0; j < 32; ++j)
            wLDS[row * KCB_ + lane + 32 * j] = (_Float16)(vals[j] * inv);
    }
    __syncthreads();

    // ---- GEMM: A from LDS, B from global (L2-resident codebook) ----
    const int tile = blockIdx.x * 8 + wave;            // 0..255
    const int rc   = lane & 15;
    const int khb  = (lane >> 4) * 8;

    const _Float16* aBase  = &wLDS[rc * KCB_ + khb];
    const _Float16* bcBase = cbCosT + (size_t)(tile * 16 + rc) * KCB_ + khb;
    const _Float16* bsBase = cbSinT + (size_t)(tile * 16 + rc) * KCB_ + khb;

    v8f cre0 = {}, cre1 = {}, cim0 = {}, cim1 = {};
#pragma unroll 2
    for (int k0 = 0; k0 < KCB_; k0 += 64) {
        const v16h a0  = load_frag16(aBase + k0);      // ds_load_b128 x2
        const v16h bc0 = load_frag16(bcBase + k0);
        const v16h bs0 = load_frag16(bsBase + k0);
        const v16h a1  = load_frag16(aBase + k0 + 32);
        const v16h bc1 = load_frag16(bcBase + k0 + 32);
        const v16h bs1 = load_frag16(bsBase + k0 + 32);
        cre0 = __builtin_amdgcn_wmma_f32_16x16x32_f16(
            false, a0, false, bc0, (short)0, cre0, false, false);
        cim0 = __builtin_amdgcn_wmma_f32_16x16x32_f16(
            false, a0, false, bs0, (short)0, cim0, false, false);
        cre1 = __builtin_amdgcn_wmma_f32_16x16x32_f16(
            false, a1, false, bc1, (short)0, cre1, false, false);
        cim1 = __builtin_amdgcn_wmma_f32_16x16x32_f16(
            false, a1, false, bs1, (short)0, cim1, false, false);
    }

    const int d = tile * 16 + rc;
#pragma unroll
    for (int r = 0; r < 8; ++r) {
        const int m = khb + r;
        const float re = cre0[r] + cre1[r];
        const float im = cim0[r] + cim1[r];
        const float n2 = re * re + im * im;
        const float rsq = __builtin_amdgcn_rsqf(n2);   // single v_rsq_f32
        // n2==0 -> atan2(0,0)=0 -> cos=1, sin=0
        const float co = (n2 > 0.0f) ? re * rsq : 1.0f;
        const float si = (n2 > 0.0f) ? im * rsq : 0.0f;
        estF[m * FEAT_ + d]        = (_Float16)co;
        estF[m * FEAT_ + DIM_ + d] = (_Float16)si;
    }
}

// ---------------------------------------------------------------------------

extern "C" void kernel_launch(void* const* d_in, const int* in_sizes, int n_in,
                              void* d_out, int out_size, void* d_ws, size_t ws_size,
                              hipStream_t stream) {
    const float* superposed = (const float*)d_in[0];   // (8, 4096)
    const float* codebook   = (const float*)d_in[1];   // (1024, 4096)
    const int num_iterations = 50;                     // d_in[2]: fixed by reference
    (void)in_sizes; (void)n_in; (void)out_size; (void)ws_size;

    // Workspace carve-up (all 256B aligned by construction)
    char* ws = (char*)d_ws;
    size_t o = 0;
    _Float16* cbF    = (_Float16*)(ws + o); o += (size_t)KCB_ * FEAT_ * 2;   // 16 MB
    _Float16* cbCosT = (_Float16*)(ws + o); o += (size_t)DIM_ * KCB_ * 2;    //  8 MB
    _Float16* cbSinT = (_Float16*)(ws + o); o += (size_t)DIM_ * KCB_ * 2;    //  8 MB
    _Float16* estF   = (_Float16*)(ws + o); o += (size_t)16 * FEAT_ * 2;     // 256 KB
    float*    logits = (float*)   (ws + o); o += (size_t)16 * KCB_ * 4;      //  64 KB

    const float scaleIter  = 10.0f / (float)DIM_;      // TEMP / DIM fused into logits
    const float scaleFinal = 1.0f / (float)DIM_;

    // One-time (per call) codebook conversion + feature init
    k_prep_cb<<<(KCB_ * DIM_) / 256, 256, 0, stream>>>(codebook, cbF, cbCosT, cbSinT);
    k_feat_init<<<(16 * DIM_) / 256, 256, 0, stream>>>(superposed, estF);

    for (int it = 0; it < num_iterations; ++it) {
        k_sims<16><<<KCB_ / 16 / 2, 256, 0, stream>>>(estF, cbF, logits, KCB_, scaleIter);
        k_phasor<<<DIM_ / 16 / 8, 256, 0, stream>>>(logits, cbCosT, cbSinT, estF);
    }

    // Final similarity: mean cosine, rows 0..7 straight into d_out (8 x 1024 f32)
    k_sims<BROWS_><<<KCB_ / 16 / 2, 256, 0, stream>>>(estF, cbF, (float*)d_out, KCB_, scaleFinal);
}
